// ILCMDecoder_23991687316171
// MI455X (gfx1250) — matrix operand order repaired and checked by reading, not compiled, not run on hardware
//
#include <hip/hip_runtime.h>

typedef __attribute__((ext_vector_type(16))) _Float16 v16h;
typedef __attribute__((ext_vector_type(8)))  _Float16 v8h;
typedef __attribute__((ext_vector_type(8)))  float    v8f;

#define NB   65536
#define DZv  32
#define PHv  5
#define HIDv 512
#define DXv  256
#define RPB  64    // rows of E per decoder block (4 M-tiles)
#define HSTR 520   // padded f16 row stride for LDS H tile (16B-aligned rows, breaks bank conflicts)
#define LOG2PI 1.8378770664093453f

// ---------------- Prepass 1: sigmoid mask folded into W1, zero flags ----------------
__global__ __launch_bounds__(256) void ilcm_prep_mask(const float* __restrict__ adj,
                                                      const float* __restrict__ W1,
                                                      float* __restrict__ W1m,
                                                      int* __restrict__ flags) {
  int tid = threadIdx.x;
  if (tid < DZv) flags[tid] = 0;
  for (int idx = tid; idx < DZv * PHv * DZv; idx += 256) {
    int n = idx / (PHv * DZv);
    int d = idx % DZv;
    float mval;
    if (d < n)      mval = 1.0f / (1.0f + expf(-adj[d * DZv + n]));        // A[d][n], upper tri
    else if (d > n) mval = 1.0f - 1.0f / (1.0f + expf(-adj[n * DZv + d])); // 1 - A[n][d]
    else            mval = 0.0f;
    W1m[idx] = W1[idx] * mval;
  }
}

// ---------------- Prepass 2: f32 -> f16 transposed weights ----------------
// Wd1t[n*32 + k] = (f16) Wd1[k*512 + n]   (= Wd1^T row-major, 512 x 32)
// Wd2t[n*512 + k] = (f16) Wd2[k*256 + n]  (= Wd2^T row-major, 256 x 512)
__global__ __launch_bounds__(256) void ilcm_prep_weights(const float* __restrict__ Wd1,
                                                         const float* __restrict__ Wd2,
                                                         _Float16* __restrict__ Wd1t,
                                                         _Float16* __restrict__ Wd2t) {
  int t = blockIdx.x * 256 + threadIdx.x;
  if (t < HIDv * DZv) {
    int n = t / DZv, k = t % DZv;
    Wd1t[t] = (_Float16)Wd1[k * HIDv + n];
  }
  if (t < HIDv * DXv) {
    int n = t / HIDv, k = t % HIDv;
    Wd2t[t] = (_Float16)Wd2[k * DXv + n];
  }
}

// ---------------- Prepass 3: active[n] = any_b(intervention[b, n+1] > 0) ----------------
__global__ __launch_bounds__(256) void ilcm_prep_active(const int* __restrict__ interv,
                                                        int* __restrict__ flags) {
  __shared__ int f[DZv];
  int tid = threadIdx.x;
  if (tid < DZv) f[tid] = 0;
  __syncthreads();
  int b = blockIdx.x * 256 + tid;
  const int* row = interv + (size_t)b * (DZv + 1);
  for (int j = 1; j <= DZv; ++j)
    if (row[j] > 0) atomicOr(&f[j - 1], 1);
  __syncthreads();
  if (tid < DZv && f[tid]) atomicOr(&flags[tid], 1);
}

// ---------------- Decoder: Out = relu(E @ Wd1 + bd1) @ Wd2 + bd2, f16 WMMA, f32 accum ----------------
// Block = 256 threads (8 waves), 64 rows (4 batch tiles) of E per block.
// Stage 1 computes the TRANSPOSED product H^T-tile = Wd1^T x E^T so each lane's
// 8 D components are contiguous hid values of one batch row -> packed
// ds_store_b128 into the row-major H that stage 2 consumes with b128 loads.
__global__ __launch_bounds__(256) void ilcm_decoder(const float* __restrict__ E,
                                                    const _Float16* __restrict__ Wd1t,
                                                    const _Float16* __restrict__ Wd2t,
                                                    const float* __restrict__ bd1,
                                                    const float* __restrict__ bd2,
                                                    float* __restrict__ Out) {
  __shared__ _Float16 Hs[RPB * HSTR];  // H tile, row-major [batch][hid], ~66.6 KB
  const int row0 = blockIdx.x * RPB;
  const int lane = threadIdx.x & 31;
  const int wv   = threadIdx.x >> 5;
  const int m    = lane & 15;
  const int hf   = lane >> 4;

  // ---- Stage-1 B operands: E^T tiles (32dz x 16batch). Lane = batch col m,
  //      components c -> K = hf*16 + c (contiguous 16 f32 = two 32B loads). ----
  v16h eb[4];
#pragma unroll
  for (int mt = 0; mt < 4; ++mt) {
    const float* erow = E + (size_t)(row0 + mt * 16 + m) * DZv;
    v8f elo = *(const v8f*)(erow + hf * 16);
    v8f ehi = *(const v8f*)(erow + hf * 16 + 8);
#pragma unroll
    for (int c = 0; c < 8; ++c) {
      eb[mt][c]     = (_Float16)elo[c];
      eb[mt][c + 8] = (_Float16)ehi[c];
    }
  }

  // ---- Stage 1: H^T tiles = Wd1^T (A) x E^T (B); 32 hid-tiles, 4 per wave ----
#pragma unroll
  for (int t = 0; t < 4; ++t) {
    const int hidb = (wv * 4 + t) * 16;
    // A fragment: row M = hid = hidb + m; K runs hf*16+c
    const _Float16* wrow = Wd1t + (hidb + m) * DZv;
    v8h wlo = *(const v8h*)(wrow + hf * 16);
    v8h whi = *(const v8h*)(wrow + hf * 16 + 8);
    v16h afrag;
#pragma unroll
    for (int c = 0; c < 8; ++c) { afrag[c] = wlo[c]; afrag[c + 8] = whi[c]; }
    // bias per D component: M = hid = hidb + hf*8 + r (aligned 32B vector load)
    v8f bias = *(const v8f*)(bd1 + hidb + hf * 8);
#pragma unroll
    for (int mt = 0; mt < 4; ++mt) {
      v8f acc = bias;
      acc = __builtin_amdgcn_wmma_f32_16x16x32_f16(false, afrag, false, eb[mt],
                                                   (short)0, acc, false, false);
      v8h hv;
#pragma unroll
      for (int r = 0; r < 8; ++r) hv[r] = (_Float16)__builtin_fmaxf(acc[r], 0.0f);
      // D: (M = hidb+hf*8+r, N = batch m) -> one packed 16B store, row-major H
      *(v8h*)(Hs + (mt * 16 + m) * HSTR + hidb + hf * 8) = hv;
    }
  }
  __syncthreads();

  // ---- Stage 2: Out = H @ Wd2 + bd2; 16 N-tiles, 2 per wave, K=512 (16 steps of 32) ----
  const int n0 = (wv * 2 + 0) * 16 + m;
  const int n1 = (wv * 2 + 1) * 16 + m;
  v8f acc[4][2];
#pragma unroll
  for (int mt = 0; mt < 4; ++mt)
#pragma unroll
    for (int r = 0; r < 8; ++r) { acc[mt][0][r] = bd2[n0]; acc[mt][1][r] = bd2[n1]; }

#pragma unroll 2
  for (int kk = 0; kk < 16; ++kk) {
    int kb = kk * 32;
    v16h b0 = *(const v16h*)(Wd2t + n0 * HIDv + kb + hf * 16);
    v16h b1 = *(const v16h*)(Wd2t + n1 * HIDv + kb + hf * 16);
#pragma unroll
    for (int mt = 0; mt < 4; ++mt) {
      // A fragment = two aligned 16B LDS loads (K runs contiguous per lane-half)
      const _Float16* hrow = Hs + (mt * 16 + m) * HSTR;
      v8h lo = *(const v8h*)(hrow + kb + hf * 16);      // components 0..7
      v8h hi = *(const v8h*)(hrow + kb + hf * 16 + 8);  // components 8..15
      v16h af;
#pragma unroll
      for (int r = 0; r < 8; ++r) { af[r] = lo[r]; af[r + 8] = hi[r]; }
      acc[mt][0] = __builtin_amdgcn_wmma_f32_16x16x32_f16(false, af, false, b0,
                                                          (short)0, acc[mt][0], false, false);
      acc[mt][1] = __builtin_amdgcn_wmma_f32_16x16x32_f16(false, af, false, b1,
                                                          (short)0, acc[mt][1], false, false);
    }
  }

  float* obase = Out + (size_t)row0 * DXv;
#pragma unroll
  for (int mt = 0; mt < 4; ++mt) {
#pragma unroll
    for (int r = 0; r < 8; ++r) {
      int orow = (mt * 16 + hf * 8 + r) * DXv;
      obase[orow + n0] = acc[mt][0][r];
      obase[orow + n1] = acc[mt][1][r];
    }
  }
}

// ---------------- log_p kernel: one thread per sample; weights staged in LDS ----------------
__global__ __launch_bounds__(256) void ilcm_logp(const float* __restrict__ e1,
                                                 const float* __restrict__ e2,
                                                 const float* __restrict__ W1m,
                                                 const float* __restrict__ W2,
                                                 const float* __restrict__ b1,
                                                 const float* __restrict__ b2,
                                                 const float* __restrict__ W3,
                                                 const float* __restrict__ b3,
                                                 const int* __restrict__ flags,
                                                 float* __restrict__ partials) {
  __shared__ float sW1m[DZv * PHv * DZv];  // 5120
  __shared__ float sW2[DZv * PHv * PHv];   // 800
  __shared__ float sW3[DZv * 2 * PHv];     // 320 (only d=0,1 of W3 are used)
  __shared__ float sb1[DZv * PHv], sb2[DZv * PHv];
  __shared__ float sb3[DZv * 2];
  __shared__ float sact[DZv];
  __shared__ float red[256];

  int tid = threadIdx.x;
  for (int i = tid; i < DZv * PHv * DZv; i += 256) sW1m[i] = W1m[i];
  for (int i = tid; i < DZv * PHv * PHv; i += 256) sW2[i] = W2[i];
  for (int i = tid; i < DZv * 2 * PHv; i += 256) {
    int n = i / (2 * PHv);
    int r = i % (2 * PHv);
    int d = r / PHv, mi = r % PHv;
    sW3[i] = W3[(n * DZv + d) * PHv + mi];
  }
  for (int i = tid; i < DZv * PHv; i += 256) { sb1[i] = b1[i]; sb2[i] = b2[i]; }
  for (int i = tid; i < DZv * 2; i += 256) { sb3[i] = b3[(i / 2) * DZv + (i % 2)]; }
  if (tid < DZv) sact[tid] = flags[tid] ? 1.0f : 0.0f;
  __syncthreads();

  int b = blockIdx.x * 256 + tid;
  const float* r1 = e1 + (size_t)b * DZv;
  const float* r2 = e2 + (size_t)b * DZv;
  float x1[DZv];
  float local = 0.0f;
#pragma unroll
  for (int d = 0; d < DZv; ++d) {
    x1[d] = r1[d];
    local += -0.5f * x1[d] * x1[d];  // log_p_e1 data part (const added at end)
  }

  for (int n = 0; n < DZv; ++n) {
    float h1[PHv];
#pragma unroll
    for (int k = 0; k < PHv; ++k) {
      float s = sb1[n * PHv + k];
      const float* w = &sW1m[(n * PHv + k) * DZv];
#pragma unroll
      for (int d = 0; d < DZv; ++d) s += w[d] * x1[d];
      h1[k] = s > 0.0f ? s : 0.0f;
    }
    float h2[PHv];
#pragma unroll
    for (int mi = 0; mi < PHv; ++mi) {
      float s = sb2[n * PHv + mi];
#pragma unroll
      for (int k = 0; k < PHv; ++k) s += sW2[(n * PHv + mi) * PHv + k] * h1[k];
      h2[mi] = s > 0.0f ? s : 0.0f;
    }
    float shift = sb3[n * 2 + 0];
    float ls    = sb3[n * 2 + 1];
#pragma unroll
    for (int mi = 0; mi < PHv; ++mi) {
      shift += sW3[n * 2 * PHv + mi] * h2[mi];
      ls    += sW3[n * 2 * PHv + PHv + mi] * h2[mi];
    }
    float z = (r2[n] - shift) * expf(-ls);
    local += sact[n] * (-0.5f * z * z - 0.5f * LOG2PI - ls);
  }

  red[tid] = local;
  __syncthreads();
  for (int s = 128; s > 0; s >>= 1) {  // fixed-order tree: deterministic
    if (tid < s) red[tid] += red[tid + s];
    __syncthreads();
  }
  if (tid == 0) partials[blockIdx.x] = red[0];
}

// ---------------- Final deterministic reduction + constants ----------------
__global__ void ilcm_final(const float* __restrict__ partials, float* __restrict__ out) {
  if (threadIdx.x == 0 && blockIdx.x == 0) {
    float s = 0.0f;
    for (int i = 0; i < NB / 256; ++i) s += partials[i];
    s += -0.5f * LOG2PI * (float)(NB * DZv);        // constant part of log_p_e1
    s += -logf((float)(DZv + 1)) * (float)NB;       // log_p_I
    out[0] = s;
  }
}

extern "C" void kernel_launch(void* const* d_in, const int* in_sizes, int n_in,
                              void* d_out, int out_size, void* d_ws, size_t ws_size,
                              hipStream_t stream) {
  const float* e1   = (const float*)d_in[0];
  const float* e2   = (const float*)d_in[1];
  const float* adj  = (const float*)d_in[2];
  const float* W1   = (const float*)d_in[3];
  const float* b1   = (const float*)d_in[4];
  const float* W2   = (const float*)d_in[5];
  const float* b2   = (const float*)d_in[6];
  const float* W3   = (const float*)d_in[7];
  const float* b3   = (const float*)d_in[8];
  const float* Wd1  = (const float*)d_in[9];
  const float* bd1  = (const float*)d_in[10];
  const float* Wd2  = (const float*)d_in[11];
  const float* bd2  = (const float*)d_in[12];
  const int* interv = (const int*)d_in[13];
  float* out = (float*)d_out;

  // workspace layout (all offsets 32B-aligned where vector loads occur)
  char* ws = (char*)d_ws;
  float*    W1m      = (float*)(ws + 0);        // 5120 f32  -> 20480 B
  int*      flags    = (int*)(ws + 20480);      // 32 i32    -> 20608 B
  float*    partials = (float*)(ws + 20608);    // 256 f32   -> 21632 B
  _Float16* Wd1t     = (_Float16*)(ws + 21760); // 16384 f16 -> 54528 B
  _Float16* Wd2t     = (_Float16*)(ws + 54784); // 131072 f16-> 316928 B

  ilcm_prep_mask<<<1, 256, 0, stream>>>(adj, W1, W1m, flags);
  ilcm_prep_weights<<<(HIDv * DXv) / 256, 256, 0, stream>>>(Wd1, Wd2, Wd1t, Wd2t);
  ilcm_prep_active<<<NB / 256, 256, 0, stream>>>(interv, flags);

  ilcm_decoder<<<NB / RPB, 256, 0, stream>>>(e1, Wd1t, Wd2t, bd1, bd2, out);
  ilcm_decoder<<<NB / RPB, 256, 0, stream>>>(e2, Wd1t, Wd2t, bd1, bd2,
                                             out + (size_t)NB * DXv);

  ilcm_logp<<<NB / 256, 256, 0, stream>>>(e1, e2, W1m, W2, b1, b2, W3, b3,
                                          flags, partials);
  ilcm_final<<<1, 64, 0, stream>>>(partials, out + 2 * (size_t)NB * DXv);
}
